// VectorQuantizer_75582834475424
// MI455X (gfx1250) — matrix operand order repaired and checked by reading, not compiled, not run on hardware
//
#include <hip/hip_runtime.h>

typedef __attribute__((ext_vector_type(16))) _Float16 v16h;
typedef __attribute__((ext_vector_type(8)))  float    v8f;
typedef __attribute__((ext_vector_type(4)))  unsigned int v4u;
typedef __attribute__((ext_vector_type(8)))  int          v8i;
typedef __attribute__((ext_vector_type(4)))  int          v4i;

#define N_ROWS        262144      // 64 * 4096
#define K_CODES       1024
#define DIMS          64
#define CHUNK         256         // codebook rows per LDS buffer
#define NCHUNK        (K_CODES / CHUNK)
#define CHUNK_BYTES   (CHUNK * DIMS * 2)               // 32 KB of f16
#define ROWS_PER_WAVE 16
#define WAVES_PER_WG  8
#define ROWS_PER_WG   (ROWS_PER_WAVE * WAVES_PER_WG)   // 128
#define NUM_WGS       (N_ROWS / ROWS_PER_WG)           // 2048

// ---------------------------------------------------------------------------
// Prep kernel: emb f32 -> f16 (once, into d_ws) + scale[k] = -0.5 / sum(e^2)
// ---------------------------------------------------------------------------
__global__ __launch_bounds__(256)
void vq_prep_kernel(const float* __restrict__ emb,
                    _Float16* __restrict__ ws_h,
                    float* __restrict__ ws_scl)
{
    const int r = blockIdx.x * 256 + threadIdx.x;     // one codebook row per thread
    const float4* src = (const float4*)(emb + (size_t)r * DIMS);
    _Float16* dst = ws_h + (size_t)r * DIMS;
    float s2 = 0.f;
#pragma unroll
    for (int q = 0; q < DIMS / 4; ++q) {
        const float4 v = src[q];
        s2 += v.x * v.x + v.y * v.y + v.z * v.z + v.w * v.w;
        union { _Float16 h[4]; uint2 u; } p;
        p.h[0] = (_Float16)v.x; p.h[1] = (_Float16)v.y;
        p.h[2] = (_Float16)v.z; p.h[3] = (_Float16)v.w;
        *(uint2*)(dst + 4 * q) = p.u;
    }
    ws_scl[r] = -0.5f / s2;
}

// ---------------------------------------------------------------------------
// TDM: 1-D contiguous copy global -> LDS (32 KB chunk), data_size = 8 bytes
// ---------------------------------------------------------------------------
__device__ __forceinline__ void tdm_load_chunk(const _Float16* gsrc, _Float16* lds_dst)
{
    const unsigned int       lds_off = (unsigned int)(unsigned long long)lds_dst;
    const unsigned long long ga      = (unsigned long long)gsrc;
    const unsigned int nunits = CHUNK_BYTES / 8;      // 4096 x 8-byte elements

    v4u g0;
    g0[0] = 1u;                                           // count=1, user mode
    g0[1] = lds_off;                                      // lds_addr   (bits 63:32)
    g0[2] = (unsigned int)(ga & 0xFFFFFFFFu);             // global_addr[31:0]
    g0[3] = (unsigned int)((ga >> 32) & 0x1FFFFFFu)       // global_addr[56:32]
          | (2u << 30);                                   // type = 2 ("image")

    v8i g1;
    g1[0] = (int)(3u << 16);                              // wg_mask=0, data_size=3 (8B)
    g1[1] = (int)((nunits & 0xFFFFu) << 16);              // tensor_dim0[15:0]
    g1[2] = (int)((nunits >> 16) & 0xFFFFu);              // tensor_dim0[31:16], dim1=0
    g1[3] = (int)((nunits & 0xFFFFu) << 16);              // tile_dim0
    g1[4] = 0;                                            // tile_dim1/2 unused
    g1[5] = (int)nunits;                                  // tensor_dim0_stride lo
    g1[6] = 0;
    g1[7] = 0;

    const v4i z4 = {0, 0, 0, 0};
    const v8i z8 = {0, 0, 0, 0, 0, 0, 0, 0};
    // clang-23 / therock-10.0 form: 6 args (extra int32x8 group before cpol)
    __builtin_amdgcn_tensor_load_to_lds(g0, g1, z4, z4, z8, 0);
}

// ---------------------------------------------------------------------------
// Main kernel: WMMA distance scan + argmin + gather + loss
// ---------------------------------------------------------------------------
__global__ __launch_bounds__(256)
void vq_wmma_kernel(const float* __restrict__ z,
                    const float* __restrict__ emb,
                    const _Float16* __restrict__ ws_h,
                    const float* __restrict__ ws_scl,
                    float* __restrict__ out_zq,
                    float* __restrict__ out_idx,
                    float* __restrict__ out_loss)
{
    // double-buffered f16 codebook chunks, filled by the Tensor Data Mover
    __shared__ __align__(32) _Float16 lds_emb[2][CHUNK * DIMS];   // 2 x 32 KB

    const int tid  = threadIdx.x;
    const int lane = tid & 31;
    const int wave = tid >> 5;
    const int j    = lane & 15;      // column / row-in-tile selector
    const int g    = lane >> 4;      // lane group (0 or 1)
    const int base_row = (blockIdx.x * WAVES_PER_WG + wave) * ROWS_PER_WAVE;

    // ---- A fragments: 16 z-rows, f32 -> f16 (contiguous k = 16g+e order, same as B) ----
    const float* zrow = z + (size_t)(base_row + j) * DIMS;
    v16h a0, a1;
    float ssq = 0.f;
#pragma unroll
    for (int s = 0; s < 2; ++s) {
#pragma unroll
        for (int p = 0; p < 8; ++p) {
            const int e = 2 * p;
            const int K = 32 * s + 16 * g + e;
            const float2 v = *(const float2*)(zrow + K);
            const _Float16 h0 = (_Float16)v.x;
            const _Float16 h1 = (_Float16)v.y;
            if (s == 0) { a0[e] = h0; a0[e + 1] = h1; }
            else        { a1[e] = h0; a1[e + 1] = h1; }
            ssq += v.x * v.x + v.y * v.y;
        }
    }
    ssq += __shfl_xor(ssq, 16);           // lane groups hold disjoint row halves
    float zs[8];
#pragma unroll
    for (int r = 0; r < 8; ++r) zs[r] = __shfl(ssq, r + 8 * g);

    float best[8];
    int   bidx[8];
#pragma unroll
    for (int r = 0; r < 8; ++r) { best[r] = __builtin_inff(); bidx[r] = 0; }

    // kick off DMA of chunk 0 (wave-uniform guard -> single TDM issue)
    if (wave == 0) tdm_load_chunk(ws_h, &lds_emb[0][0]);

    for (int chunk = 0; chunk < NCHUNK; ++chunk) {
        if (wave == 0) __builtin_amdgcn_s_wait_tensorcnt(0);
        __syncthreads();                  // publish chunk's LDS buffer to all waves
        if (wave == 0 && chunk + 1 < NCHUNK)
            tdm_load_chunk(ws_h + (size_t)(chunk + 1) * CHUNK * DIMS,
                           &lds_emb[(chunk + 1) & 1][0]);   // overlap DMA w/ compute

        const _Float16* buf = &lds_emb[chunk & 1][0];
        const float* sclp = ws_scl + chunk * CHUNK;

#pragma unroll 2
        for (int cl = 0; cl < CHUNK / 16; ++cl) {
            // B 32x16 f16 fragments: lane (g,j) supplies col j, K = 16g+e (+32)
            const v16h b0 = *(const v16h*)&buf[(cl * 16 + j) * DIMS + g * 16];
            const v16h b1 = *(const v16h*)&buf[(cl * 16 + j) * DIMS + 32 + g * 16];
            const float scl = sclp[cl * 16 + j];            // -0.5/e_sq (L2-resident)

            v8f acc = {};
            acc = __builtin_amdgcn_wmma_f32_16x16x32_f16(false, a0, false, b0,
                                                         (short)0, acc, false, false);
            acc = __builtin_amdgcn_wmma_f32_16x16x32_f16(false, a1, false, b1,
                                                         (short)0, acc, false, false);

            const int col = chunk * CHUNK + cl * 16 + j;
#pragma unroll
            for (int r = 0; r < 8; ++r) {
                // distances = (z_sq / e_sq / -2) * dot
                const float dist = (zs[r] * scl) * acc[r];
                if (dist < best[r]) { best[r] = dist; bidx[r] = col; }
            }
        }
        // next iteration's wait+barrier protects buffer reuse
    }

    // ---- argmin reduction across the 16 col-lanes (first-index tie-break) --
#pragma unroll
    for (int r = 0; r < 8; ++r) {
#pragma unroll
        for (int m = 1; m <= 8; m <<= 1) {
            const float ov = __shfl_xor(best[r], m);
            const int   oi = __shfl_xor(bidx[r], m);
            if (ov < best[r] || (ov == best[r] && oi < bidx[r])) {
                best[r] = ov; bidx[r] = oi;
            }
        }
    }

    if (j < 8) out_idx[base_row + 8 * g + j] = (float)bidx[j];

    // ---- z_q gather, straight-through output, loss -------------------------
    float lsum = 0.f;
#pragma unroll
    for (int r = 0; r < 8; ++r) {
        const int row = base_row + r + 8 * g;
        const int idx = bidx[r];
        const float4 ev = *(const float4*)(emb + (size_t)idx * DIMS + 4 * j);
        const float4 zv = *(const float4*)(z   + (size_t)row * DIMS + 4 * j);
        float4 o;
        o.x = zv.x + (ev.x - zv.x);
        o.y = zv.y + (ev.y - zv.y);
        o.z = zv.z + (ev.z - zv.z);
        o.w = zv.w + (ev.w - zv.w);
        *(float4*)(out_zq + (size_t)row * DIMS + 4 * j) = o;
        const float dx = ev.x - zv.x, dy = ev.y - zv.y;
        const float dz = ev.z - zv.z, dw = ev.w - zv.w;
        lsum += dx * dx + dy * dy + dz * dz + dw * dw;
    }
#pragma unroll
    for (int m = 1; m <= 16; m <<= 1) lsum += __shfl_xor(lsum, m);
    if (lane == 0) {
        // loss = (1 + beta) * mean(diff^2), beta = 0.25
        atomicAdd(out_loss, lsum * (1.25f / (float)(N_ROWS * (size_t)DIMS)));
    }
}

extern "C" void kernel_launch(void* const* d_in, const int* in_sizes, int n_in,
                              void* d_out, int out_size, void* d_ws, size_t ws_size,
                              hipStream_t stream) {
    const float* z   = (const float*)d_in[0];   // [64,4096,64] f32
    const float* emb = (const float*)d_in[1];   // [1024,64]   f32

    float* out      = (float*)d_out;
    float* out_zq   = out;                                   // N*D floats
    float* out_idx  = out + (size_t)N_ROWS * DIMS;           // N floats
    float* out_loss = out_idx + N_ROWS;                      // 1 float

    // workspace: f16 codebook (128 KB) + scale table (4 KB)
    _Float16* ws_h   = (_Float16*)d_ws;
    float*    ws_scl = (float*)((char*)d_ws + (size_t)K_CODES * DIMS * 2);

    (void)hipMemsetAsync(out_loss, 0, sizeof(float), stream);   // capture-safe

    vq_prep_kernel<<<K_CODES / 256, 256, 0, stream>>>(emb, ws_h, ws_scl);
    vq_wmma_kernel<<<NUM_WGS, 256, 0, stream>>>(z, emb, ws_h, ws_scl,
                                                out_zq, out_idx, out_loss);
}
